// GATSingleHead_68281390072319
// MI455X (gfx1250) — compile-verified, matrix operand order checked by reading, not run on hardware
//
#include <hip/hip_runtime.h>
#include <hip/hip_bf16.h>
#include <math.h>

typedef __attribute__((ext_vector_type(16))) _Float16 v16h;
typedef __attribute__((ext_vector_type(8)))  _Float16 v8h;
typedef __attribute__((ext_vector_type(8)))  float    v8f;
typedef __attribute__((ext_vector_type(4)))  float    f32x4;

#define ED 16
#define NEG_SLOPE 0.2f

// ---------------- utility kernels ----------------

__global__ void zero_f32(float* __restrict__ p, long n) {
    long i = (long)blockIdx.x * blockDim.x + threadIdx.x;
    if (i < n) p[i] = 0.0f;
}

__device__ __forceinline__ void atomicMaxF(float* addr, float val) {
    int* ai = (int*)addr;
    int old = __hip_atomic_load(ai, __ATOMIC_RELAXED, __HIP_MEMORY_SCOPE_AGENT);
    while (__int_as_float(old) < val) {
        int assumed = old;
        old = atomicCAS(ai, assumed, __float_as_int(val));
        if (old == assumed) break;
    }
}

// ---------------- self-loop edge_attr = segment-mean ----------------

__global__ void accum_loop_attr(const int* __restrict__ dst,
                                const float* __restrict__ eattr,
                                float* __restrict__ cnt,
                                float* __restrict__ lsum, long E) {
    long e = (long)blockIdx.x * blockDim.x + threadIdx.x;
    if (e >= E) return;
    int d = dst[e];
    atomicAdd(&cnt[d], 1.0f);
    const float* ea = eattr + e * ED;
    float* ls = lsum + (long)d * ED;
#pragma unroll
    for (int j = 0; j < ED; ++j) atomicAdd(&ls[j], ea[j]);
}

__global__ void finalize_loop_attr(float* __restrict__ lsum,
                                   const float* __restrict__ cnt, long n) {
    long i = (long)blockIdx.x * blockDim.x + threadIdx.x;
    if (i >= n * ED) return;
    float c = cnt[i / ED];
    lsum[i] = lsum[i] / fmaxf(c, 1.0f);
}

// ---------------- WMMA GEMM:  H[N,ncols] = X[N,K] * W[K,ncols] ----------------
// Block = 8 waves sharing one 16-wide column tile (blockIdx.y).
// Each wave owns one 16-row tile; K stepped by 32 with v_wmma_f32_16x16x32_f16.
// A (16-bit 16x32 layout): lane L holds two contiguous K-runs [8g, 8g+7] and
// [16+8g, 23+8g] -> fetched as 4x b128 f32 loads + v_cvt_pk_f16_f32.
// B (32x16): staged once per block into LDS as f16, column-major with stride
// 40 halves (80B: 16B-aligned for ds_load_b128, bank-conflict-free), then two
// ds_load_b128 per lane.  Requires nrows % 16 == 0 (holds: N = 100000).

#define LDS_STRIDE 40

__global__ void __launch_bounds__(256)
gemm16_wmma(const float* __restrict__ X, const float* __restrict__ W,
            float* __restrict__ H, int nrows, int K, int ncols) {
    __shared__ _Float16 lw[16 * LDS_STRIDE];

    const int wave = threadIdx.x >> 5;
    const int lane = threadIdx.x & 31;
    const int numRowTiles = nrows >> 4;
    int rt = blockIdx.x * 8 + wave;
    const bool active = rt < numRowTiles;
    if (!active) rt = numRowTiles - 1;           // tail waves: compute garbage,
                                                 // keep barriers + EXEC legal
    const int colbase = blockIdx.y << 4;
    const int g  = lane >> 4;
    const int lm = lane & 15;
    const int row = rt * 16 + lm;

    // LDS fill coordinates: 256 threads cover 32(k) x 16(c) as (k, 2 cols)
    const int fk = threadIdx.x >> 3;             // 0..31
    const int fc = (threadIdx.x & 7) << 1;       // 0,2,..,14

    v8f acc = {};
    for (int k0 = 0; k0 < K; k0 += 32) {
        // stage W tile (32 x 16) -> LDS f16 column-major
        float2 wv = *(const float2*)(W + (long)(k0 + fk) * ncols + colbase + fc);
        lw[fc * LDS_STRIDE + fk]       = (_Float16)wv.x;
        lw[(fc + 1) * LDS_STRIDE + fk] = (_Float16)wv.y;
        __syncthreads();

        // A: two contiguous 8-float runs per lane
        const float* ax = X + (long)row * K + k0 + 8 * g;
        f32x4 a0 = *(const f32x4*)(ax);
        f32x4 a1 = *(const f32x4*)(ax + 4);
        f32x4 a2 = *(const f32x4*)(ax + 16);
        f32x4 a3 = *(const f32x4*)(ax + 20);
        v16h a;
#pragma unroll
        for (int i = 0; i < 4; ++i) {
            a[i]      = (_Float16)a0[i];
            a[4 + i]  = (_Float16)a1[i];
            a[8 + i]  = (_Float16)a2[i];
            a[12 + i] = (_Float16)a3[i];
        }

        // B: two 16B LDS reads per lane
        v8h b0 = *(const v8h*)(lw + lm * LDS_STRIDE + 8 * g);
        v8h b1 = *(const v8h*)(lw + lm * LDS_STRIDE + 16 + 8 * g);
        v16h b;
#pragma unroll
        for (int i = 0; i < 8; ++i) { b[i] = b0[i]; b[8 + i] = b1[i]; }

        // (neg_a, A, neg_b, B, c_mod, C, reuse_a, reuse_b)
        acc = __builtin_amdgcn_wmma_f32_16x16x32_f16(false, a, false, b,
                                                     (short)0, acc, false, false);
        __syncthreads();
    }

    if (active) {                                // wave-uniform guard
        const int col = colbase + lm;
#pragma unroll
        for (int r = 0; r < 8; ++r)              // D layout: VGPR r -> row r+8g
            H[(long)(rt * 16 + r + 8 * g) * ncols + col] = acc[r];
    }
}

// ---------------- wa = We(ED x dout) @ att_e(dout) ----------------

__global__ void compute_wa(const float* __restrict__ We,
                           const float* __restrict__ atte,
                           float* __restrict__ wa, int dout) {
    int j = threadIdx.x;
    if (j >= ED) return;
    float s = 0.0f;
    for (int k = 0; k < dout; ++k) s += We[j * dout + k] * atte[k];
    wa[j] = s;
}

// ---------------- per-node attention dots + init max/den ----------------

__global__ void node_alpha(const float* __restrict__ h,
                           const float* __restrict__ att_s,
                           const float* __restrict__ att_d,
                           float* __restrict__ as_, float* __restrict__ ad_,
                           float* __restrict__ mx, float* __restrict__ den,
                           long n, int dout) {
    long v = (long)blockIdx.x * blockDim.x + threadIdx.x;
    if (v >= n) return;
    const float* hv = h + v * dout;
    float s = 0.0f, d = 0.0f;
    for (int k = 0; k < dout; ++k) { float x = hv[k]; s += x * att_s[k]; d += x * att_d[k]; }
    as_[v] = s; ad_[v] = d;
    mx[v] = -INFINITY; den[v] = 0.0f;
}

// ---------------- edge passes ----------------

__global__ void edge_alpha_max(const int* __restrict__ src, const int* __restrict__ dst,
                               const float* __restrict__ eattr,
                               const float* __restrict__ loop_attr,
                               const float* __restrict__ wa,
                               const float* __restrict__ as_, const float* __restrict__ ad_,
                               float* __restrict__ alpha, float* __restrict__ mx,
                               long E, long n) {
    long e = (long)blockIdx.x * blockDim.x + threadIdx.x;
    if (e >= E + n) return;
    int s, d; const float* ea;
    if (e < E) { s = src[e]; d = dst[e]; ea = eattr + e * ED; }
    else       { s = d = (int)(e - E);   ea = loop_attr + (long)s * ED; }
    float ae = 0.0f;
#pragma unroll
    for (int j = 0; j < ED; ++j) ae += ea[j] * wa[j];
    float a = as_[s] + ad_[d] + ae;
    a = a > 0.0f ? a : NEG_SLOPE * a;                // leaky relu
    alpha[e] = a;
    atomicMaxF(&mx[d], a);
}

__global__ void edge_exp(const int* __restrict__ dst,
                         float* __restrict__ alpha,
                         const float* __restrict__ mx, float* __restrict__ den,
                         long E, long n) {
    long e = (long)blockIdx.x * blockDim.x + threadIdx.x;
    if (e >= E + n) return;
    int d = (e < E) ? dst[e] : (int)(e - E);
    float ex = __expf(alpha[e] - mx[d]);
    alpha[e] = ex;
    atomicAdd(&den[d], ex);
}

__global__ void edge_scatter(const int* __restrict__ src, const int* __restrict__ dst,
                             const float* __restrict__ ex, const float* __restrict__ den,
                             const float* __restrict__ h, float* __restrict__ out,
                             long E, long n, int dout) {
    long e = (long)blockIdx.x * blockDim.x + threadIdx.x;
    if (e >= E + n) return;
    int s, d;
    if (e < E) { s = src[e]; d = dst[e]; } else { s = d = (int)(e - E); }
    float c = ex[e] / (den[d] + 1e-16f);
    const float* hs = h + (long)s * dout;
    float* od = out + (long)d * dout;
    for (int j = 0; j < dout; ++j) atomicAdd(&od[j], c * hs[j]);
}

__global__ void bias_act(float* __restrict__ out, const float* __restrict__ b,
                         long n, int dout, int do_elu) {
    long i = (long)blockIdx.x * blockDim.x + threadIdx.x;
    if (i >= n * dout) return;
    float v = out[i] + b[i & (dout - 1)];            // dout is a power of two
    if (do_elu) v = v > 0.0f ? v : expm1f(v);        // jax.nn.elu
    out[i] = v;
}

// ---------------- host launch ----------------

static inline int cdiv(long a, long b) { return (int)((a + b - 1) / b); }

extern "C" void kernel_launch(void* const* d_in, const int* in_sizes, int n_in,
                              void* d_out, int out_size, void* d_ws, size_t ws_size,
                              hipStream_t stream) {
    const long N = in_sizes[0] / 128;       // x: (N,128)
    const long E = in_sizes[1] / 2;         // edge_index: (2,E)
    const long E2 = E + N;                  // with self-loops

    const float* x      = (const float*)d_in[0];
    const int*   src    = (const int*)d_in[1];
    const int*   dst    = ((const int*)d_in[1]) + E;
    const float* eattr  = (const float*)d_in[2];

    // per-layer params at 3 + 6*l : W, att_src, att_dst, We, att_e, b
    const float* P[3][6];
    for (int l = 0; l < 3; ++l)
        for (int j = 0; j < 6; ++j)
            P[l][j] = (const float*)d_in[3 + 6 * l + j];
    const int din_[3]  = {128, 64, 32};
    const int dout_[3] = {64, 32, 16};

    // workspace layout (floats)
    float* ws   = (float*)d_ws;
    float* cnt  = ws;                 // N
    float* loop = cnt  + N;           // N*ED
    float* hbuf = loop + N * ED;      // N*64 (max dout)
    float* obuf = hbuf + N * 64;      // N*64 (layer1/2 outputs, reused)
    float* asrc = obuf + N * 64;      // N
    float* adst = asrc + N;           // N
    float* mx   = adst + N;           // N
    float* den  = mx   + N;           // N
    float* exb  = den  + N;           // E2
    float* wa   = exb  + E2;          // ED

    const int T = 256;

    // --- self-loop attr = mean of incoming edge attrs ---
    zero_f32<<<cdiv(N, T), T, 0, stream>>>(cnt, N);
    zero_f32<<<cdiv(N * ED, T), T, 0, stream>>>(loop, N * ED);
    accum_loop_attr<<<cdiv(E, T), T, 0, stream>>>(dst, eattr, cnt, loop, E);
    finalize_loop_attr<<<cdiv(N * ED, T), T, 0, stream>>>(loop, cnt, N);

    const float* Xin = x;
    for (int l = 0; l < 3; ++l) {
        const int din = din_[l], dout = dout_[l];
        float* outp = (l == 2) ? (float*)d_out : obuf;

        // h = Xin @ W  (WMMA; 8 row-tiles per block, blockIdx.y = col tile)
        int numRowTiles = (int)(N >> 4);
        dim3 ggrid(cdiv(numRowTiles, 8), dout >> 4);
        gemm16_wmma<<<ggrid, 256, 0, stream>>>(Xin, P[l][0], hbuf,
                                               (int)N, din, dout);
        // wa = We @ att_e
        compute_wa<<<1, 32, 0, stream>>>(P[l][3], P[l][4], wa, dout);
        // per-node dots + init segment max/den
        node_alpha<<<cdiv(N, T), T, 0, stream>>>(hbuf, P[l][1], P[l][2],
                                                 asrc, adst, mx, den, N, dout);
        // zero accumulator (safe: previous layer's input already consumed by GEMM)
        zero_f32<<<cdiv(N * dout, T), T, 0, stream>>>(outp, N * dout);
        // edge passes
        edge_alpha_max<<<cdiv(E2, T), T, 0, stream>>>(src, dst, eattr, loop, wa,
                                                      asrc, adst, exb, mx, E, N);
        edge_exp<<<cdiv(E2, T), T, 0, stream>>>(dst, exb, mx, den, E, N);
        edge_scatter<<<cdiv(E2, T), T, 0, stream>>>(src, dst, exb, den, hbuf, outp,
                                                    E, N, dout);
        bias_act<<<cdiv(N * dout, T), T, 0, stream>>>(outp, P[l][5], N, dout,
                                                      l < 2 ? 1 : 0);
        Xin = outp;
    }
}
